// GptOssDFlashAttention_22952305230270
// MI455X (gfx1250) — matrix operand order repaired and checked by reading, not compiled, not run on hardware
//
#include <hip/hip_runtime.h>

#define NHEADS   64
#define KVHEADS  8
#define HD       64
#define QLEN     1024
#define CTXLEN   1024
#define SLEN     2048
#define HDIM     2880
#define QKDIM    4096   /* NHEADS*HD */
#define KVDIM    512    /* KVHEADS*HD */
#define ATT_SCALE 0.125f
#define LOG2E    1.4426950408889634f

#ifndef __has_builtin
#define __has_builtin(x) 0
#endif
#if __has_builtin(__builtin_amdgcn_tensor_load_to_lds)
#define USE_TDM 1
#else
#define USE_TDM 0
#endif

typedef __attribute__((ext_vector_type(16))) __bf16 v16bf;
typedef __attribute__((ext_vector_type(8)))  __bf16 v8bf;
typedef __attribute__((ext_vector_type(4)))  __bf16 v4bf;
typedef __attribute__((ext_vector_type(8)))  float  v8f;
typedef __attribute__((ext_vector_type(4)))  float  f4;
typedef __attribute__((ext_vector_type(4)))  unsigned int u32x4;
typedef __attribute__((ext_vector_type(8)))  int i32x8;
typedef __attribute__((ext_vector_type(4)))  int i32x4;

static __device__ inline v16bf mk16(v8bf lo, v8bf hi) {
  return __builtin_shufflevector(lo, hi, 0,1,2,3,4,5,6,7,8,9,10,11,12,13,14,15);
}
static __device__ inline float rmax16(float v) {
  v = fmaxf(v, __shfl_xor(v, 1, 32));
  v = fmaxf(v, __shfl_xor(v, 2, 32));
  v = fmaxf(v, __shfl_xor(v, 4, 32));
  v = fmaxf(v, __shfl_xor(v, 8, 32));
  return v;
}
static __device__ inline float rsum16(float v) {
  v += __shfl_xor(v, 1, 32);
  v += __shfl_xor(v, 2, 32);
  v += __shfl_xor(v, 4, 32);
  v += __shfl_xor(v, 8, 32);
  return v;
}

#if USE_TDM
// Build a 2D Tensor-DMA descriptor (D# groups 0/1 per ISA 08_async_tensor
// §8.3/8.4) and issue TENSOR_LOAD_TO_LDS: tile_rows x tile_cols bf16 tile
// whose rows are spaced row_len elements apart in global memory.
// This toolchain exposes the 6-arg builtin: (g0, g1, g2, g3, g4, cpol).
static __device__ __attribute__((always_inline)) void
tdm_load_tile(const __bf16* gsrc, unsigned lds_off, int row_len,
              int nrows_tensor, int tile_cols, int tile_rows) {
  unsigned long long ga = (unsigned long long)(size_t)gsrc;
  u32x4 g0;
  g0[0] = 1u;                                   // count=1, user mode, no gather
  g0[1] = lds_off;                              // lds_addr (bytes)
  g0[2] = (unsigned)ga;                         // global_addr[31:0]
  g0[3] = (unsigned)((ga >> 32) & 0x01FFFFFFu) | (2u << 30);  // addr[56:32]|type=2
  i32x8 g1;
  g1[0] = 1 << 16;                              // wg_mask=0, data_size=1 (2 bytes)
  g1[1] = (int)(((unsigned)row_len & 0xFFFFu) << 16);           // tensor_dim0 lo16
  g1[2] = (int)((((unsigned)row_len >> 16) & 0xFFFFu) |
                (((unsigned)nrows_tensor & 0xFFFFu) << 16));    // dim0 hi | dim1 lo
  g1[3] = (int)((((unsigned)nrows_tensor >> 16) & 0xFFFFu) |
                ((unsigned)tile_cols << 16));                   // dim1 hi | tile_dim0
  g1[4] = tile_rows & 0xFFFF;                   // tile_dim1 | tile_dim2=0
  g1[5] = row_len;                              // tensor_dim0_stride[31:0]
  g1[6] = 0;                                    // stride hi | dim1_stride (2D unused)
  g1[7] = 0;
  i32x4 z4 = {0, 0, 0, 0};                      // D# groups 2/3: unused (2D tile)
  i32x8 z8 = {0, 0, 0, 0, 0, 0, 0, 0};
  __builtin_amdgcn_tensor_load_to_lds(g0, g1, z4, z4, z8, 0);
}
#endif

// ---------------------------------------------------------------------------
// GEMM: C[M][N] = A[M][K](bf16) * W[N][K](bf16)^T + bias[N], f32 out.
// 256 threads (8 waves), block tile 256x64, BK=64; each wave owns 32x64.
// Double-buffered LDS staged by the Tensor Data Mover (wave 0 issues DMA,
// waits TENSORcnt, barrier releases the workgroup).
// ---------------------------------------------------------------------------
#define GBM 256
#define GBN 64
#define GBK 64

__global__ __launch_bounds__(256)
void gemm_tdm_bf16(const __bf16* __restrict__ A, const __bf16* __restrict__ W,
                   const float* __restrict__ bias, float* __restrict__ C,
                   int M, int N, int K) {
  __shared__ __align__(32) __bf16 As[2][GBM][GBK];   // 64 KB
  __shared__ __align__(32) __bf16 Ws[2][GBN][GBK];   // 16 KB
  const int tid  = threadIdx.x;
  const int lane = tid & 31, wave = tid >> 5;
  const int m = lane & 15, hi = lane >> 4;
  const int rowblock = blockIdx.y * GBM;
  const int colblock = blockIdx.x * GBN;
  const int nk = K / GBK;
  v8f acc[2][4] = {};

  auto compute_step = [&](int buf) {
    const __bf16 (*at)[GBK] = As[buf];
    const __bf16 (*wt)[GBK] = Ws[buf];
#pragma unroll
    for (int ks = 0; ks < GBK; ks += 32) {
      v16bf af[2];
#pragma unroll
      for (int g = 0; g < 2; ++g) {
        v8bf lo  = *(const v8bf*)&at[wave * 32 + g * 16 + m][ks + hi * 8];
        v8bf hi8 = *(const v8bf*)&at[wave * 32 + g * 16 + m][ks + hi * 8 + 16];
        af[g] = mk16(lo, hi8);
      }
#pragma unroll
      for (int j = 0; j < 4; ++j) {
        v16bf wf = mk16(*(const v8bf*)&wt[j * 16 + m][ks + hi * 8],
                        *(const v8bf*)&wt[j * 16 + m][ks + hi * 8 + 16]);
#pragma unroll
        for (int g = 0; g < 2; ++g)
          acc[g][j] = __builtin_amdgcn_wmma_f32_16x16x32_bf16(
              false, af[g], false, wf, (short)0, acc[g][j], false, false);
      }
    }
  };

#if USE_TDM
  if (wave == 0) {
    tdm_load_tile(A + (size_t)rowblock * K, (unsigned)(size_t)&As[0][0][0],
                  K, M, GBK, GBM);
    tdm_load_tile(W + (size_t)colblock * K, (unsigned)(size_t)&Ws[0][0][0],
                  K, N, GBK, GBN);
  }
  for (int i = 0; i < nk; ++i) {
    const int buf = i & 1;
    if (wave == 0) __builtin_amdgcn_s_wait_tensorcnt(0);  // tile i landed in LDS
    __syncthreads();                                      // everyone sees it;
                                                          // buf^1 readers done
    if (wave == 0 && i + 1 < nk) {                        // prefetch tile i+1 via TDM
      const int k0 = (i + 1) * GBK;
      tdm_load_tile(A + (size_t)rowblock * K + k0,
                    (unsigned)(size_t)&As[buf ^ 1][0][0], K, M, GBK, GBM);
      tdm_load_tile(W + (size_t)colblock * K + k0,
                    (unsigned)(size_t)&Ws[buf ^ 1][0][0], K, N, GBK, GBN);
    }
    compute_step(buf);
  }
#else
  for (int i = 0; i < nk; ++i) {
    const int k0 = i * GBK;
    for (int t = tid; t < (GBM * GBK) / 8; t += 256) {
      int r = t >> 3, c = (t & 7) * 8;
      *(v8bf*)&As[0][r][c] = *(const v8bf*)(A + (size_t)(rowblock + r) * K + k0 + c);
    }
    for (int t = tid; t < (GBN * GBK) / 8; t += 256) {
      int r = t >> 3, c = (t & 7) * 8;
      *(v8bf*)&Ws[0][r][c] = *(const v8bf*)(W + (size_t)(colblock + r) * K + k0 + c);
    }
    __syncthreads();
    compute_step(0);
    __syncthreads();
  }
#endif

#pragma unroll
  for (int g = 0; g < 2; ++g)
#pragma unroll
    for (int j = 0; j < 4; ++j) {
      int col = colblock + j * 16 + m;
      float b = bias[col];
#pragma unroll
      for (int r = 0; r < 8; ++r) {
        int row = rowblock + wave * 32 + g * 16 + r + 8 * hi;
        C[(size_t)row * N + col] = acc[g][j][r] + b;
      }
    }
}

// ---------------------------------------------------------------------------
// f32 -> bf16 streaming convert (4 elements / thread)
// ---------------------------------------------------------------------------
__global__ void cvt_bf16_kernel(const float* __restrict__ src,
                                __bf16* __restrict__ dst, int n4) {
  int i = blockIdx.x * blockDim.x + threadIdx.x;
  if (i >= n4) return;
  f4 x = *(const f4*)(src + (size_t)i * 4);
  v4bf y;
#pragma unroll
  for (int u = 0; u < 4; ++u) y[u] = (__bf16)x[u];
  *(v4bf*)(dst + (size_t)i * 4) = y;
}

// ---------------------------------------------------------------------------
// RoPE + pack kernels (f32 proj outputs -> bf16 attention layouts)
// ---------------------------------------------------------------------------
__global__ void rope_pack_q_kernel(const float* __restrict__ qf,
                                   const float* __restrict__ cosb,
                                   const float* __restrict__ sinb,
                                   __bf16* __restrict__ qo) {
  int idx = blockIdx.x * blockDim.x + threadIdx.x;
  if (idx >= QLEN * QKDIM) return;
  int d = idx & (HD - 1);
  int h = (idx >> 6) & (NHEADS - 1);
  int p = idx >> 12;
  size_t base = (size_t)p * QKDIM + (size_t)h * HD;
  float v   = qf[base + d];
  float rot = (d < HD / 2) ? -qf[base + d + HD / 2] : qf[base + d - HD / 2];
  float c = cosb[(size_t)(CTXLEN + p) * HD + d];
  float s = sinb[(size_t)(CTXLEN + p) * HD + d];
  qo[((size_t)h * QLEN + p) * HD + d] = (__bf16)(v * c + rot * s);
}

__global__ void rope_pack_k_kernel(const float* __restrict__ kf,
                                   const float* __restrict__ cosb,
                                   const float* __restrict__ sinb,
                                   __bf16* __restrict__ ko) {
  int idx = blockIdx.x * blockDim.x + threadIdx.x;
  if (idx >= SLEN * KVDIM) return;
  int d = idx & (HD - 1);
  int h = (idx >> 6) & (KVHEADS - 1);
  int p = idx >> 9;
  size_t base = (size_t)p * KVDIM + (size_t)h * HD;
  float v   = kf[base + d];
  float rot = (d < HD / 2) ? -kf[base + d + HD / 2] : kf[base + d - HD / 2];
  float c = cosb[(size_t)p * HD + d];
  float s = sinb[(size_t)p * HD + d];
  ko[((size_t)h * SLEN + p) * HD + d] = (__bf16)(v * c + rot * s);
}

__global__ void pack_v_kernel(const float* __restrict__ vf,
                              __bf16* __restrict__ vt) {
  int idx = blockIdx.x * blockDim.x + threadIdx.x;
  if (idx >= SLEN * KVDIM) return;
  int d = idx & (HD - 1);
  int h = (idx >> 6) & (KVHEADS - 1);
  int p = idx >> 9;
  // transpose: vt[kvh][d][s]  -> PV B-fragments are contiguous loads
  vt[((size_t)h * HD + d) * SLEN + p] = (__bf16)vf[(size_t)p * KVDIM + h * HD + d];
}

// ---------------------------------------------------------------------------
// Flash attention: grid (NHEADS, QLEN/64), 128 threads = 4 waves; each wave
// owns a 16x64 output tile; online softmax over 32-key chunks; bf16 output.
// ---------------------------------------------------------------------------
__global__ __launch_bounds__(128)
void attn_wmma(const __bf16* __restrict__ Qb, const __bf16* __restrict__ Kb,
               const __bf16* __restrict__ VTb, __bf16* __restrict__ O) {
  __shared__ __align__(32) __bf16 plds[4][16][32];
  const int lane = threadIdx.x & 31, wave = threadIdx.x >> 5;
  const int m = lane & 15, hi = lane >> 4;
  const int head  = blockIdx.x;
  const int qbase = blockIdx.y * 64 + wave * 16;
  const int kvh   = head >> 3;   // N_REP = 8
  const __bf16* Qh = Qb  + ((size_t)head * QLEN + qbase) * HD;
  const __bf16* Kh = Kb  + (size_t)kvh * SLEN * HD;
  const __bf16* Vh = VTb + (size_t)kvh * HD * SLEN;

  v16bf qa[2];
#pragma unroll
  for (int j = 0; j < 2; ++j) {
    v8bf lo  = *(const v8bf*)(Qh + (size_t)m * HD + j * 32 + hi * 8);
    v8bf hi8 = *(const v8bf*)(Qh + (size_t)m * HD + j * 32 + hi * 8 + 16);
    qa[j] = mk16(lo, hi8);
  }

  float mrow[8], lrow[8];
#pragma unroll
  for (int r = 0; r < 8; ++r) { mrow[r] = -1e30f; lrow[r] = 0.f; }
  v8f acc[4] = {};

  for (int kb = 0; kb < SLEN; kb += 32) {
    // ---- S = Q K^T ----
    v8f s[2];
#pragma unroll
    for (int t = 0; t < 2; ++t) {
      const __bf16* krow = Kh + (size_t)(kb + t * 16 + m) * HD;
      v16bf b0 = mk16(*(const v8bf*)(krow + hi * 8), *(const v8bf*)(krow + hi * 8 + 16));
      v16bf b1 = mk16(*(const v8bf*)(krow + 32 + hi * 8), *(const v8bf*)(krow + 32 + hi * 8 + 16));
      v8f z = {};
      z    = __builtin_amdgcn_wmma_f32_16x16x32_bf16(false, qa[0], false, b0,
                                                     (short)0, z, false, false);
      s[t] = __builtin_amdgcn_wmma_f32_16x16x32_bf16(false, qa[1], false, b1,
                                                     (short)0, z, false, false);
    }
    // ---- online softmax ----
#pragma unroll
    for (int r = 0; r < 8; ++r) {
      float s0 = s[0][r] * ATT_SCALE;
      float s1 = s[1][r] * ATT_SCALE;
      float v  = rmax16(fmaxf(s0, s1));
      float mnew = fmaxf(mrow[r], v);
      float corr = exp2f((mrow[r] - mnew) * LOG2E);
      mrow[r] = mnew;
      lrow[r] *= corr;
#pragma unroll
      for (int j = 0; j < 4; ++j) acc[j][r] *= corr;
      float p0 = exp2f((s0 - mnew) * LOG2E);
      float p1 = exp2f((s1 - mnew) * LOG2E);
      lrow[r] += rsum16(p0 + p1);
      plds[wave][r + 8 * hi][m]      = (__bf16)p0;
      plds[wave][r + 8 * hi][16 + m] = (__bf16)p1;
    }
    asm volatile("s_wait_dscnt 0" ::: "memory");  // cross-lane LDS RAW, same wave
    v8bf plo = *(const v8bf*)&plds[wave][m][hi * 8];
    v8bf phi = *(const v8bf*)&plds[wave][m][hi * 8 + 16];
    v16bf pa = mk16(plo, phi);
#pragma unroll
    for (int j = 0; j < 4; ++j) {
      v16bf vb = mk16(*(const v8bf*)(Vh + (size_t)(j * 16 + m) * SLEN + kb + hi * 8),
                      *(const v8bf*)(Vh + (size_t)(j * 16 + m) * SLEN + kb + hi * 8 + 16));
      acc[j] = __builtin_amdgcn_wmma_f32_16x16x32_bf16(
          false, pa, false, vb, (short)0, acc[j], false, false);
    }
  }
#pragma unroll
  for (int r = 0; r < 8; ++r) {
    float inv = 1.0f / lrow[r];
    int row = qbase + r + 8 * hi;
#pragma unroll
    for (int j = 0; j < 4; ++j) {
      int col = head * HD + j * 16 + m;
      O[(size_t)row * QKDIM + col] = (__bf16)(acc[j][r] * inv);
    }
  }
}

// ---------------------------------------------------------------------------
extern "C" void kernel_launch(void* const* d_in, const int* in_sizes, int n_in,
                              void* d_out, int out_size, void* d_ws, size_t ws_size,
                              hipStream_t stream) {
  (void)in_sizes; (void)n_in; (void)out_size; (void)ws_size;
  const float* hidden = (const float*)d_in[0];
  const float* target = (const float*)d_in[1];
  const float* cosb   = (const float*)d_in[2];
  const float* sinb   = (const float*)d_in[3];
  const float* Wq = (const float*)d_in[4];
  const float* bq = (const float*)d_in[5];
  const float* Wk = (const float*)d_in[6];
  const float* bk = (const float*)d_in[7];
  const float* Wv = (const float*)d_in[8];
  const float* bv = (const float*)d_in[9];
  const float* Wo = (const float*)d_in[10];
  const float* bo = (const float*)d_in[11];

  char* ws = (char*)d_ws;
  __bf16* hbf  = (__bf16*)(ws);                      // 5.90 MB hidden bf16
  __bf16* tbf  = (__bf16*)(ws + 5898240u);           // 5.90 MB target bf16
  __bf16* wbig = (__bf16*)(ws + 11796480u);          // 23.6 MB Wq bf16, later Wo bf16
  __bf16* wkb  = (__bf16*)(ws + 35389440u);          // 2.95 MB
  __bf16* wvb  = (__bf16*)(ws + 38338560u);          // 2.95 MB
  float*  qf32 = (float*)(ws + 41287680u);           // 16.8 MB (reused as attn bf16)
  float*  kf32 = (float*)(ws + 58064896u);           // 4.19 MB
  float*  vf32 = (float*)(ws + 62259200u);           // 4.19 MB
  __bf16* qbf  = (__bf16*)(ws + 66453504u);          // 8.39 MB
  __bf16* kbf  = (__bf16*)(ws + 74842112u);          // 2.10 MB
  __bf16* vbt  = (__bf16*)(ws + 76939264u);          // 2.10 MB
  __bf16* attnbf = (__bf16*)(ws + 41287680u);        // reuse qf32 region

  // --- convert operands to bf16 (streaming) ---
  auto cvt = [&](const float* s, __bf16* d, int n) {
    cvt_bf16_kernel<<<(n / 4 + 255) / 256, 256, 0, stream>>>(s, d, n / 4);
  };
  cvt(hidden, hbf, QLEN * HDIM);
  cvt(target, tbf, CTXLEN * HDIM);
  cvt(Wq, wbig, QKDIM * HDIM);
  cvt(Wk, wkb, KVDIM * HDIM);
  cvt(Wv, wvb, KVDIM * HDIM);

  // --- projections (TDM-fed bf16 WMMA GEMMs) ---
  dim3 blk(256);
  gemm_tdm_bf16<<<dim3(QKDIM / GBN, QLEN / GBM), blk, 0, stream>>>(
      hbf, wbig, bq, qf32, QLEN, QKDIM, HDIM);
  gemm_tdm_bf16<<<dim3(KVDIM / GBN, QLEN / GBM), blk, 0, stream>>>(
      tbf, wkb, bk, kf32, QLEN, KVDIM, HDIM);
  gemm_tdm_bf16<<<dim3(KVDIM / GBN, QLEN / GBM), blk, 0, stream>>>(
      hbf, wkb, bk, kf32 + (size_t)CTXLEN * KVDIM, QLEN, KVDIM, HDIM);
  gemm_tdm_bf16<<<dim3(KVDIM / GBN, QLEN / GBM), blk, 0, stream>>>(
      tbf, wvb, bv, vf32, QLEN, KVDIM, HDIM);
  gemm_tdm_bf16<<<dim3(KVDIM / GBN, QLEN / GBM), blk, 0, stream>>>(
      hbf, wvb, bv, vf32 + (size_t)CTXLEN * KVDIM, QLEN, KVDIM, HDIM);

  // Wq bf16 dead after Q-proj -> stage Wo bf16 into the same buffer
  cvt(Wo, wbig, HDIM * QKDIM);

  // --- RoPE + bf16 repack ---
  rope_pack_q_kernel<<<(QLEN * QKDIM + 255) / 256, 256, 0, stream>>>(qf32, cosb, sinb, qbf);
  rope_pack_k_kernel<<<(SLEN * KVDIM + 255) / 256, 256, 0, stream>>>(kf32, cosb, sinb, kbf);
  pack_v_kernel<<<(SLEN * KVDIM + 255) / 256, 256, 0, stream>>>(vf32, vbt);

  // --- flash attention (bf16 out, overwrites dead qf32 region) ---
  attn_wmma<<<dim3(NHEADS, QLEN / 64), dim3(128), 0, stream>>>(qbf, kbf, vbt, attnbf);

  // --- output projection straight into d_out (f32) ---
  gemm_tdm_bf16<<<dim3(HDIM / GBN, QLEN / GBM), blk, 0, stream>>>(
      attnbf, wbig, bo, (float*)d_out, QLEN, HDIM, QKDIM);
}
